// GATLayer_20169166422461
// MI455X (gfx1250) — matrix operand order repaired and checked by reading, not compiled, not run on hardware
//
#include <hip/hip_runtime.h>
#include <hip/hip_bf16.h>
#include <math.h>

typedef __attribute__((ext_vector_type(2)))  float        v2f;
typedef __attribute__((ext_vector_type(8)))  float        v8f;
typedef __attribute__((ext_vector_type(16))) __bf16       v16bf;
typedef __attribute__((ext_vector_type(4)))  unsigned int v4u;
typedef __attribute__((ext_vector_type(8)))  int          v8i;
typedef __attribute__((ext_vector_type(4)))  int          v4i;

#define B_    8
#define N_    2048
#define FIN   256
#define FOUT  64
#define ALPHA 0.2f
#define NEGINF -9000000000000000.0f

// ---------------------------------------------------------------------------
// Kernel 1: Wh = h @ W  (f32 WMMA 16x16x4), fused e_src = Wh.a1, e_dst = Wh.a2
// Stores Wh transposed in bf16: whbt[(b*FOUT + o)*N_ + j]  (B operand for k2).
// One wave (32 threads) per 16-row tile.
// ---------------------------------------------------------------------------
__global__ __launch_bounds__(32) void gat_wh_kernel(
    const float* __restrict__ h, const float* __restrict__ W,
    const float* __restrict__ a, __hip_bfloat16* __restrict__ whbt,
    float* __restrict__ esrc, float* __restrict__ edst)
{
    const int tile = blockIdx.x;              // B_*N_/16 tiles
    const int b    = tile / (N_ / 16);
    const int i0   = (tile % (N_ / 16)) * 16;
    const int lane = threadIdx.x;             // 0..31
    const int m    = lane & 15;               // A/B minor index
    const int hl   = lane >> 4;               // lane half

    v8f acc[4] = {};                          // 16 rows x 64 cols (4 n-tiles)

    const float* hrow = h + (size_t)(b * N_ + i0 + m) * FIN;
    for (int k0 = 0; k0 < FIN; k0 += 4) {
        // A 16x4 f32: lanes 0-15 -> K = k0+0,k0+1 ; lanes 16-31 -> K = k0+2,k0+3
        const int ka = k0 + 2 * hl;
        v2f av;
        av[0] = hrow[ka];
        av[1] = hrow[ka + 1];
#pragma unroll
        for (int t = 0; t < 4; t++) {
            const int n = 16 * t + m;
            v2f bv;
            bv[0] = W[ka * FOUT + n];
            bv[1] = W[(ka + 1) * FOUT + n];
            acc[t] = __builtin_amdgcn_wmma_f32_16x16x4_f32(
                false, av, false, bv, (short)0, acc[t], false, false);
        }
    }

    // e_src / e_dst: dot each output row with a1 / a2 (cross-lane reduce).
    const float* a1 = a;
    const float* a2 = a + FOUT;
    float ps[8], pd[8];
#pragma unroll
    for (int r = 0; r < 8; r++) { ps[r] = 0.f; pd[r] = 0.f; }
#pragma unroll
    for (int t = 0; t < 4; t++) {
        const int n = 16 * t + m;
        const float w1 = a1[n], w2 = a2[n];
#pragma unroll
        for (int r = 0; r < 8; r++) {
            ps[r] += acc[t][r] * w1;
            pd[r] += acc[t][r] * w2;
        }
    }
#pragma unroll
    for (int r = 0; r < 8; r++) {
        for (int ofs = 1; ofs < 16; ofs <<= 1) {
            ps[r] += __shfl_xor(ps[r], ofs, 32);
            pd[r] += __shfl_xor(pd[r], ofs, 32);
        }
    }
    if (m == 0) {
#pragma unroll
        for (int r = 0; r < 8; r++) {
            const int row = i0 + r + 8 * hl;   // C layout: vgpr r -> row r + 8*half
            esrc[b * N_ + row] = ps[r];
            edst[b * N_ + row] = pd[r];
        }
    }

    // bf16 transposed Wh for kernel 2's contiguous B-operand loads.
#pragma unroll
    for (int t = 0; t < 4; t++) {
        const int n = 16 * t + m;
#pragma unroll
        for (int r = 0; r < 8; r++) {
            const int row = i0 + r + 8 * hl;
            whbt[(size_t)(b * FOUT + n) * N_ + row] = __float2bfloat16(acc[t][r]);
        }
    }
}

// ---------------------------------------------------------------------------
// TDM: DMA a 16x64 int32 tile of adj (row stride N_) into LDS.
// D# per CDNA5 ISA ch.8: group0 = {count/type/addr}, group1 = dims/strides,
// groups 2/3 zero (2D tile).  Tracked with TENSORcnt.
// ---------------------------------------------------------------------------
__device__ __forceinline__ void tdm_load_adj_tile(const int* gaddr, unsigned lds_off)
{
    const unsigned long long ga = (unsigned long long)(uintptr_t)gaddr;
    v4u g0;
    g0[0] = 1u;                                              // count=1, user mode
    g0[1] = lds_off;                                         // lds_addr (bytes)
    g0[2] = (unsigned)(ga & 0xFFFFFFFFu);                    // global_addr[31:0]
    g0[3] = (unsigned)((ga >> 32) & 0x01FFFFFFu)             // global_addr[56:32]
          | 0x80000000u;                                     // type = 2 ("image")
    v8i g1;
    g1[0] = (int)(2u << 16);                                 // data_size = 4 bytes
    g1[1] = (int)((unsigned)(N_ & 0xFFFF) << 16);            // tensor_dim0[15:0]
    g1[2] = (int)((unsigned)(N_ >> 16)                       // tensor_dim0[31:16]
          | ((unsigned)(N_ & 0xFFFF) << 16));                // tensor_dim1[15:0]
    g1[3] = (int)((unsigned)(N_ >> 16) | (64u << 16));       // tensor_dim1 hi | tile_dim0=64
    g1[4] = 16;                                              // tile_dim1=16, tile_dim2=0
    g1[5] = N_;                                              // tensor_dim0_stride[31:0]
    g1[6] = 0;                                               // stride hi | dim1_stride lo
    g1[7] = 0;
    v4i g2 = {};
    v4i g3 = {};
    v8i g4 = {};
    __builtin_amdgcn_tensor_load_to_lds(g0, g1, g2, g3, g4, 0);
}

// ---------------------------------------------------------------------------
// Kernel 2: fused mask + online softmax + attention @ Wh (bf16 WMMA 16x16x32)
// 4 waves per 16-row tile, each owning a quarter of the j range.  adj tiles
// are DMA'd into LDS by the Tensor Data Mover, double buffered, synchronized
// with s_wait_tensorcnt.  Partials merged through LDS at the end; ELU + store.
// ---------------------------------------------------------------------------
__global__ __launch_bounds__(128) void gat_attn_kernel(
    const int* __restrict__ adj, const __hip_bfloat16* __restrict__ whbt,
    const float* __restrict__ esrc, const float* __restrict__ edst,
    float* __restrict__ out)
{
    __shared__ int   adjt[4][2][16][64];  // [wave][buf][row][j] : TDM destination
    __shared__ float accs[4][4][8][32];   // [wave][ntile][reg][lane]
    __shared__ float mrow[4][16];
    __shared__ float lrow[4][16];

    const int tile = blockIdx.x;
    const int b    = tile / (N_ / 16);
    const int i0   = (tile % (N_ / 16)) * 16;
    const int lane = threadIdx.x & 31;
    const int wave = threadIdx.x >> 5;
    const int m    = lane & 15;
    const int hl   = lane >> 4;

    const int jbeg = wave * (N_ / 4);
    const int jend = jbeg + (N_ / 4);

    const float src_m = esrc[b * N_ + i0 + m];                 // row m logit half
    const float* edb  = edst + b * N_;
    const int* gbase  = adj + (size_t)(b * N_ + i0) * N_;      // tile row 0

    const unsigned lds0 = (unsigned)(uintptr_t)&adjt[wave][0][0][0];
    const unsigned lds1 = (unsigned)(uintptr_t)&adjt[wave][1][0][0];

    float mr = -INFINITY;   // running row max   (row = m, replicated both halves)
    float lr = 0.f;         // running row sumexp
    v8f acc[4] = {};        // f32 accumulators, C layout rows r + 8*hl

    // Prime the pipeline: DMA first 16x64 adj tile.
    tdm_load_adj_tile(gbase + jbeg, lds0);

    int c = 0;
    for (int jc = jbeg; jc < jend; jc += 64, c++) {
        const int nb = c & 1;
        if (jc + 64 < jend) {
            tdm_load_adj_tile(gbase + (jc + 64), nb ? lds0 : lds1);
            __builtin_amdgcn_s_wait_tensorcnt(1);   // current tile landed (in-order)
        } else {
            __builtin_amdgcn_s_wait_tensorcnt(0);
        }
        const int* at = &adjt[wave][nb][m][0];

#pragma unroll
        for (int s = 0; s < 2; s++) {
            const int j0 = jc + 32 * s;
            // Build the 16 A-operand values of this lane (contiguous k order, same as B's loads):
            // vgpr v: K = 16*hl + 2v {+0,+1}
            float e[16];
            float cmax = -INFINITY;
#pragma unroll
            for (int v = 0; v < 8; v++) {
                const int kb = 16 * hl + 2 * v;
                const int j  = j0 + kb;
                const int   a0  = at[32 * s + kb];
                const int   a1i = at[32 * s + kb + 1];
                const float d0 = edb[j], d1 = edb[j + 1];
                float x0 = src_m + d0; x0 = x0 > 0.f ? x0 : ALPHA * x0;
                float x1 = src_m + d1; x1 = x1 > 0.f ? x1 : ALPHA * x1;
                e[2 * v]     = (a0  > 0) ? x0 : NEGINF;
                e[2 * v + 1] = (a1i > 0) ? x1 : NEGINF;
                cmax = fmaxf(cmax, fmaxf(e[2 * v], e[2 * v + 1]));
            }
            cmax = fmaxf(cmax, __shfl_xor(cmax, 16, 32));      // combine halves
            const float mnew  = fmaxf(mr, cmax);
            const float scale = __expf(mr - mnew);             // exp(-inf)=0 first iter

            float lsum = 0.f;
            v16bf avec;
#pragma unroll
            for (int ee = 0; ee < 16; ee++) {
                const float p = __expf(e[ee] - mnew);
                lsum += p;
                avec[ee] = (__bf16)p;
            }
            lsum += __shfl_xor(lsum, 16, 32);
            lr = lr * scale + lsum;
            mr = mnew;

            // Rescale accumulators; acc reg r holds row r + 8*hl.
#pragma unroll
            for (int r = 0; r < 8; r++) {
                const float sc = __shfl(scale, r + 8 * hl, 32);
#pragma unroll
                for (int t = 0; t < 4; t++) acc[t][r] *= sc;
            }

            // B operand: WhbT rows are contiguous in j -> one aligned 32B load.
#pragma unroll
            for (int t = 0; t < 4; t++) {
                const int n = 16 * t + m;
                const __hip_bfloat16* bp =
                    whbt + (size_t)(b * FOUT + n) * N_ + j0 + 16 * hl;
                const v16bf bvec = *(const v16bf*)bp;
                acc[t] = __builtin_amdgcn_wmma_f32_16x16x32_bf16(
                    false, avec, false, bvec, (short)0, acc[t], false, false);
            }
        }
    }

    // Publish per-wave partials.
#pragma unroll
    for (int t = 0; t < 4; t++)
#pragma unroll
        for (int r = 0; r < 8; r++)
            accs[wave][t][r][lane] = acc[t][r];
    if (hl == 0) { mrow[wave][m] = mr; lrow[wave][m] = lr; }
    __syncthreads();

    // Wave 0 merges the 4 softmax partitions, normalizes, applies ELU, stores.
    if (wave == 0) {
#pragma unroll
        for (int r = 0; r < 8; r++) {
            const int row = r + 8 * hl;
            const float mM = fmaxf(fmaxf(mrow[0][row], mrow[1][row]),
                                   fmaxf(mrow[2][row], mrow[3][row]));
            float sc[4];
            float lt = 0.f;
#pragma unroll
            for (int w = 0; w < 4; w++) {
                sc[w] = __expf(mrow[w][row] - mM);
                lt += lrow[w][row] * sc[w];
            }
            const float inv = 1.0f / lt;
#pragma unroll
            for (int t = 0; t < 4; t++) {
                float s = 0.f;
#pragma unroll
                for (int w = 0; w < 4; w++) s += accs[w][t][r][lane] * sc[w];
                s *= inv;
                const float o = s > 0.f ? s : __expf(s) - 1.0f;
                const int n = 16 * t + m;
                out[(size_t)(b * N_ + i0 + row) * FOUT + n] = o;
            }
        }
    }
}

// ---------------------------------------------------------------------------
extern "C" void kernel_launch(void* const* d_in, const int* in_sizes, int n_in,
                              void* d_out, int out_size, void* d_ws, size_t ws_size,
                              hipStream_t stream)
{
    const float* h   = (const float*)d_in[0];
    const int*   adj = (const int*)d_in[1];
    const float* W   = (const float*)d_in[2];
    const float* a   = (const float*)d_in[3];
    float*       out = (float*)d_out;

    char* ws = (char*)d_ws;
    __hip_bfloat16* whbt = (__hip_bfloat16*)ws;                 // 8*64*2048*2 = 2 MB
    float* esrc = (float*)(ws + (size_t)B_ * FOUT * N_ * 2);    // 64 KB
    float* edst = esrc + B_ * N_;                               // 64 KB

    dim3 g1(B_ * N_ / 16), b1(32);
    gat_wh_kernel<<<g1, b1, 0, stream>>>(h, W, a, whbt, esrc, edst);

    dim3 g2(B_ * N_ / 16), b2(128);
    gat_attn_kernel<<<g2, b2, 0, stream>>>(adj, whbt, esrc, edst, out);
}
